// CVKAN_30751965839658
// MI455X (gfx1250) — matrix operand order repaired and checked by reading, not compiled, not run on hardware
//
#include <hip/hip_runtime.h>
#include <hip/hip_bf16.h>

// CVKAN fused 2-layer kernel for gfx1250 (MI455X).
// GEMM-reformulated: A = [RBF basis | silu features], computed on the fly into
// LDS; weights streamed to f16 LDS tiles (double-buffered, one barrier per
// K-block); v_wmma_f32_16x16x32_f16 accumulate. Layer-0 output h kept in LDS.

typedef __attribute__((ext_vector_type(16))) _Float16 v16h;
typedef __attribute__((ext_vector_type(8)))  _Float16 v8h;
typedef __attribute__((ext_vector_type(8)))  float    v8f;

constexpr int B_ROWS = 8192;   // batch
constexpr int D_IN   = 128;    // input channels for BOTH layers (D0 == D1 == 128)
constexpr int BM     = 32;     // batch rows per workgroup
constexpr int LDA    = 72;     // f16 elements per A-tile row (64 + 8 pad)
constexpr int LDB    = 72;     // f16 elements per B-tile column (64 + 8 pad)
constexpr int HLD    = 136;    // f16 elements per h row (128 + 8 pad)

__device__ __forceinline__ float gridpt(int u) {
    return -2.0f + (4.0f / 7.0f) * (float)u;   // linspace(-2, 2, 8)
}

__device__ __forceinline__ float silu_f(float x) {
    return x / (1.0f + __expf(-x));
}

// A fragment: 16x32 f16, ISA layout: lanes 0-15 row M=lm (g=0), 16-31 same rows g=1;
// halves 0..7 -> K = kbase+8g .. +7 ; halves 8..15 -> K = kbase+16+8g .. +7
__device__ __forceinline__ v16h load_afrag(const _Float16* sA, int m0, int kbase,
                                           int lm, int lg) {
    const _Float16* p = sA + (m0 + lm) * LDA + kbase + 8 * lg;
    v8h lo = *(const v8h*)(p);
    v8h hi = *(const v8h*)(p + 16);
    v16h r;
#pragma unroll
    for (int j = 0; j < 8; ++j) { r[j] = lo[j]; r[j + 8] = hi[j]; }
    return r;
}

// B fragment: 32x16 f16, lane (g, n) holds column c0+n, K = kbase+8g .. +7, kbase+16+8g .. +7
// LDS B tile stored [col][k] (k contiguous) so this is 2x b128 per lane.
__device__ __forceinline__ v16h load_bfrag(const _Float16* sBc, int c0, int kbase,
                                           int lm, int lg) {
    const _Float16* p = sBc + (c0 + lm) * LDB + kbase + 8 * lg;
    v8h lo = *(const v8h*)(p);
    v8h hi = *(const v8h*)(p + 16);
    v16h r;
#pragma unroll
    for (int j = 0; j < 8; ++j) { r[j] = lo[j]; r[j + 8] = hi[j]; }
    return r;
}

// One CVKAN layer over BM rows. O = output channels. K = 128*64 grid + 256 silu.
// IN_LDS: read x from sH (f16) instead of global. OUT_LDS: write h to sH else global.
template <int O, bool IN_LDS, bool OUT_LDS>
__device__ void layer_pass(const float* __restrict__ gxr, const float* __restrict__ gxi,
                           const float* __restrict__ Wr,  const float* __restrict__ Wi,
                           const float* __restrict__ swr, const float* __restrict__ swi,
                           const float* __restrict__ sbr, const float* __restrict__ sbi,
                           _Float16* sA0, _Float16* sA1, _Float16* sB0, _Float16* sB1,
                           _Float16* sH, float* sBias,
                           float* __restrict__ gOut, int row0) {
    const int tid  = threadIdx.x;
    const int lane = tid & 31;
    const int wave = tid >> 5;
    const int mt   = wave & 1;   // M tile (2 x 16 rows)
    const int ng   = wave >> 1;  // 4 column groups
    const int lg   = lane >> 4;
    const int lm   = lane & 15;
    constexpr int CT = O / 64;   // 16-col tiles per wave (layer0: 2, layer1: 1)

    v8f acc[CT][2];
#pragma unroll
    for (int ct = 0; ct < CT; ++ct)
#pragma unroll
        for (int cp = 0; cp < 2; ++cp)
            acc[ct][cp] = (v8f){0.f, 0.f, 0.f, 0.f, 0.f, 0.f, 0.f, 0.f};

    // Per-column bias: sum_i sb[i, o]  (consumed at epilogue; syncs occur in loop)
    if (tid < O) {
        float s0 = 0.f, s1 = 0.f;
        for (int i = 0; i < D_IN; ++i) {
            s0 += sbr[(size_t)i * O + tid];
            s1 += sbi[(size_t)i * O + tid];
        }
        sBias[tid]       = s0;
        sBias[128 + tid] = s1;
    }

    auto inval = [&](int row, int ch, int comp) -> float {
        if constexpr (IN_LDS) {
            return (float)sH[(comp * BM + row) * HLD + ch];
        } else {
            const float* p = comp ? gxi : gxr;
            return p[(size_t)(row0 + row) * D_IN + ch];
        }
    };

    // ---- tile fills (write the "next" buffer) ----
    auto fill_grid = [&](int i, _Float16* dA, _Float16* dB) {
        {   // A tile: P[row][u*8+v] = exp(-(xr-g_u)^2) * exp(-(xi-g_v)^2)
            const int row = tid >> 3;
            const int u   = tid & 7;
            const float xr = inval(row, i, 0);
            const float xi = inval(row, i, 1);
            const float du = xr - gridpt(u);
            const float bu = __expf(-du * du);
            v8h prod;
#pragma unroll
            for (int v = 0; v < 8; ++v) {
                const float dv = xi - gridpt(v);
                prod[v] = (_Float16)(bu * __expf(-dv * dv));
            }
            *(v8h*)(dA + row * LDA + 8 * u) = prod;
        }
        // B tile: W[i, col, u, v] -> f16 LDS [comp][col][k]
        for (int c = tid; c < 2 * O * 4; c += 256) {
            const int comp = c / (O * 4);
            const int rem  = c % (O * 4);
            const int col  = rem >> 2;
            const int kq   = (rem & 3) * 16;
            const float* src = (comp ? Wi : Wr) + ((size_t)i * O + col) * 64 + kq;
            __builtin_prefetch((const void*)(src + (size_t)O * 64), 0, 1);  // next channel
            const float4* s4 = (const float4*)src;
            const float4 f0 = s4[0], f1 = s4[1], f2 = s4[2], f3 = s4[3];
            v8h h0 = {(_Float16)f0.x, (_Float16)f0.y, (_Float16)f0.z, (_Float16)f0.w,
                      (_Float16)f1.x, (_Float16)f1.y, (_Float16)f1.z, (_Float16)f1.w};
            v8h h1 = {(_Float16)f2.x, (_Float16)f2.y, (_Float16)f2.z, (_Float16)f2.w,
                      (_Float16)f3.x, (_Float16)f3.y, (_Float16)f3.z, (_Float16)f3.w};
            _Float16* dst = dB + (size_t)comp * (O * LDB) + col * LDB + kq;
            *(v8h*)dst       = h0;
            *(v8h*)(dst + 8) = h1;
        }
    };

    // silu K-extension block j (k = j*64 .. j*64+63 of [silu(xr) | silu(xi)])
    auto fill_silu = [&](int j, _Float16* dA, _Float16* dB) {
        {
            const int row = tid >> 3;
            const int q   = tid & 7;
            v8h outv;
#pragma unroll
            for (int t = 0; t < 8; ++t) {
                const int kc = j * 64 + q * 8 + t;
                const float x = (kc < D_IN) ? inval(row, kc, 0) : inval(row, kc - D_IN, 1);
                outv[t] = (_Float16)silu_f(x);
            }
            *(v8h*)(dA + row * LDA + 8 * q) = outv;
        }
        for (int c = tid; c < 2 * O * 4; c += 256) {
            const int comp = c / (O * 4);
            const int rem  = c % (O * 4);
            const int col  = rem >> 2;
            const int kq   = (rem & 3) * 16;
            v8h h0, h1;
#pragma unroll
            for (int t = 0; t < 16; ++t) {
                const int kc = j * 64 + kq + t;
                float v;
                if (comp == 0)
                    v = (kc < D_IN) ? swr[(size_t)kc * O + col]
                                    : -swi[(size_t)(kc - D_IN) * O + col];
                else
                    v = (kc < D_IN) ? swi[(size_t)kc * O + col]
                                    :  swr[(size_t)(kc - D_IN) * O + col];
                if (t < 8) h0[t] = (_Float16)v; else h1[t - 8] = (_Float16)v;
            }
            _Float16* dst = dB + (size_t)comp * (O * LDB) + col * LDB + kq;
            *(v8h*)dst       = h0;
            *(v8h*)(dst + 8) = h1;
        }
    };

    // ---- compute: load ALL fragments for a K-step, then WMMA burst ----
    auto do_compute = [&](const _Float16* bA, const _Float16* bB) {
#pragma unroll
        for (int ks = 0; ks < 2; ++ks) {
            v16h a = load_afrag(bA, mt * 16, ks * 32, lm, lg);
            v16h bf[CT][2];
#pragma unroll
            for (int ct = 0; ct < CT; ++ct) {
                const int c0 = ng * (O / 4) + ct * 16;
#pragma unroll
                for (int cp = 0; cp < 2; ++cp)
                    bf[ct][cp] = load_bfrag(bB + (size_t)cp * (O * LDB), c0, ks * 32, lm, lg);
            }
#pragma unroll
            for (int ct = 0; ct < CT; ++ct)
#pragma unroll
                for (int cp = 0; cp < 2; ++cp)
                    acc[ct][cp] = __builtin_amdgcn_wmma_f32_16x16x32_f16(
                        false, a, false, bf[ct][cp], (short)0, acc[ct][cp], false, false);
        }
    };

    // ---- software-pipelined main loop: fill block n+1 while computing block n ----
    fill_grid(0, sA0, sB0);
    __syncthreads();

    for (int i = 0; i < D_IN; ++i) {
        _Float16* cA = (i & 1) ? sA1 : sA0;
        _Float16* cB = (i & 1) ? sB1 : sB0;
        _Float16* nA = (i & 1) ? sA0 : sA1;
        _Float16* nB = (i & 1) ? sB0 : sB1;
        if (i + 1 < D_IN) fill_grid(i + 1, nA, nB);
        else              fill_silu(0, nA, nB);
        do_compute(cA, cB);
        __syncthreads();
    }
    for (int j = 0; j < 4; ++j) {   // D_IN even -> parity continues as j & 1
        _Float16* cA = (j & 1) ? sA1 : sA0;
        _Float16* cB = (j & 1) ? sB1 : sB0;
        _Float16* nA = (j & 1) ? sA0 : sA1;
        _Float16* nB = (j & 1) ? sB0 : sB1;
        if (j + 1 < 4) fill_silu(j + 1, nA, nB);
        do_compute(cA, cB);
        __syncthreads();
    }

    // ---- epilogue: add bias, write h (LDS f16) or final output (global f32) ----
    // C layout: lanes 0-15 -> N = lm, M = vgpr; lanes 16-31 -> M = vgpr + 8
#pragma unroll
    for (int ct = 0; ct < CT; ++ct) {
        const int col = ng * (O / 4) + ct * 16 + lm;
#pragma unroll
        for (int cp = 0; cp < 2; ++cp) {
            const v8f c = acc[ct][cp];
#pragma unroll
            for (int r = 0; r < 8; ++r) {
                const int m = mt * 16 + r + 8 * lg;
                const float v = c[r] + sBias[cp * 128 + col];
                if constexpr (OUT_LDS) {
                    sH[(cp * BM + m) * HLD + col] = (_Float16)v;
                } else {
                    gOut[(size_t)(row0 + m) * (2 * O) + 2 * col + cp] = v;
                }
            }
        }
    }
}

__global__ __launch_bounds__(256)
void cvkan_fused(const float* __restrict__ xr,  const float* __restrict__ xi,
                 const float* __restrict__ w0r, const float* __restrict__ w0i,
                 const float* __restrict__ sw0r, const float* __restrict__ sw0i,
                 const float* __restrict__ sb0r, const float* __restrict__ sb0i,
                 const float* __restrict__ w1r, const float* __restrict__ w1i,
                 const float* __restrict__ sw1r, const float* __restrict__ sw1i,
                 const float* __restrict__ sb1r, const float* __restrict__ sb1i,
                 float* __restrict__ out) {
    __shared__ __align__(16) _Float16 sA0[BM * LDA];        //  4608 B
    __shared__ __align__(16) _Float16 sA1[BM * LDA];        //  4608 B
    __shared__ __align__(16) _Float16 sB0[2 * 128 * LDB];   // 36864 B
    __shared__ __align__(16) _Float16 sB1[2 * 128 * LDB];   // 36864 B
    __shared__ __align__(16) _Float16 sH[2 * BM * HLD];     // 17408 B
    __shared__ __align__(16) float    sBias[2 * 128];       //  1024 B  (~100 KB total)

    const int row0 = blockIdx.x * BM;

    // Layer 0: x (global f32) -> h (LDS f16), O = 128
    layer_pass<128, false, true>(xr, xi, w0r, w0i, sw0r, sw0i, sb0r, sb0i,
                                 sA0, sA1, sB0, sB1, sH, sBias, nullptr, row0);
    __syncthreads();
    // Layer 1: h (LDS) -> out (global f32, [B, 64, 2]), O = 64
    layer_pass<64, true, false>(nullptr, nullptr, w1r, w1i, sw1r, sw1i, sb1r, sb1i,
                                sA0, sA1, sB0, sB1, sH, sBias, out, row0);
}

extern "C" void kernel_launch(void* const* d_in, const int* in_sizes, int n_in,
                              void* d_out, int out_size, void* d_ws, size_t ws_size,
                              hipStream_t stream) {
    (void)in_sizes; (void)n_in; (void)out_size; (void)d_ws; (void)ws_size;
    const float* xr   = (const float*)d_in[0];
    const float* xi   = (const float*)d_in[1];
    const float* w0r  = (const float*)d_in[2];
    const float* w0i  = (const float*)d_in[3];
    const float* sw0r = (const float*)d_in[4];
    const float* sw0i = (const float*)d_in[5];
    const float* sb0r = (const float*)d_in[6];
    const float* sb0i = (const float*)d_in[7];
    const float* w1r  = (const float*)d_in[8];
    const float* w1i  = (const float*)d_in[9];
    const float* sw1r = (const float*)d_in[10];
    const float* sw1i = (const float*)d_in[11];
    const float* sb1r = (const float*)d_in[12];
    const float* sb1i = (const float*)d_in[13];
    float* out = (float*)d_out;

    cvkan_fused<<<B_ROWS / BM, 256, 0, stream>>>(
        xr, xi, w0r, w0i, sw0r, sw0i, sb0r, sb0i,
        w1r, w1i, sw1r, sw1i, sb1r, sb1i, out);
}
